// Encoder_56667798503611
// MI455X (gfx1250) — compile-verified, matrix operand order checked by reading, not run on hardware
//
#include <hip/hip_runtime.h>
#include <cstdint>

typedef __attribute__((ext_vector_type(16))) _Float16 v16h;
typedef __attribute__((ext_vector_type(8)))  float    v8f;

#define SCALE_ 0.022097086912079608f   // 1/sqrt(2048)
#define EPS_   1e-5f
#define LDSW_  40                      // padded LDS row stride in halves (80B)
#define GM_    4096                    // GEMM M (B*N tokens, padded B*W rows)
#define GN_    2048                    // GEMM N
#define GK_    2048                    // GEMM K
#define NSTEP_ (GK_ / 32)              // 64 K-steps

// ---------------------------------------------------------------------------
// f32 -> f16 convert
// ---------------------------------------------------------------------------
__global__ __launch_bounds__(256) void cvt_f16(const float* __restrict__ in,
                                               _Float16* __restrict__ out, int n)
{
    int i = blockIdx.x * 256 + threadIdx.x;
    if (i < n) out[i] = (_Float16)in[i];
}

// ---------------------------------------------------------------------------
// CDNA5 async global->LDS copy (ASYNCcnt-tracked, zero data VGPRs).
// ---------------------------------------------------------------------------
__device__ inline uint32_t lds_off32(const void* p)
{
    return (uint32_t)(uintptr_t)p;   // generic LDS ptr: addr[31:0] = LDS offset
}

__device__ inline void async_g2l_b128(uint32_t lds_off, const void* gptr)
{
    asm volatile("global_load_async_to_lds_b128 %0, %1, off"
                 :: "v"(lds_off), "v"(gptr)
                 : "memory");
}

__device__ inline void wait_async0()
{
#if __has_builtin(__builtin_amdgcn_s_wait_asynccnt)
    __builtin_amdgcn_s_wait_asynccnt(0);
#else
    asm volatile("s_wait_asynccnt 0x0" ::: "memory");
#endif
}

// ---------------------------------------------------------------------------
// 16x32 f16 fragment in the CDNA5 WMMA A/B register layout, from global.
// Lanes 0-15: row=lane,  halves 0..7 = K[0..7],  halves 8..15 = K[16..23]
// Lanes 16-31: row=lane-16, halves 0..7 = K[8..15], halves 8..15 = K[24..31]
// ---------------------------------------------------------------------------
__device__ inline v16h load_frag(const _Float16* __restrict__ base, int ld,
                                 int row0, int k0)
{
    int l  = threadIdx.x & 31;
    int r  = l & 15;
    int hi = l >> 4;
    const _Float16* p = base + (size_t)(row0 + r) * ld + (k0 + hi * 8);
    union { float4 f4[2]; v16h v; } u;
    u.f4[0] = *reinterpret_cast<const float4*>(p);
    u.f4[1] = *reinterpret_cast<const float4*>(p + 16);
    return u.v;
}

// Same fragment gather, from an LDS tile with padded row stride LDSW_.
__device__ inline v16h lds_frag(const _Float16* base, int row0)
{
    int l  = threadIdx.x & 31;
    int r  = l & 15;
    int hi = l >> 4;
    const _Float16* p = base + (row0 + r) * LDSW_ + hi * 8;
    union { float4 f4[2]; v16h v; } u;
    u.f4[0] = *reinterpret_cast<const float4*>(p);
    u.f4[1] = *reinterpret_cast<const float4*>(p + 16);
    return u.v;
}

// ---------------------------------------------------------------------------
// C[4096,2048] = A[4096,2048] * W[2048,2048]^T + bias   (f16 in, f32 out)
// block = 256 threads (8 waves), block tile 128x256, wave tile 64x64 (4x4),
// K-step 32, double-buffered LDS filled by async global->LDS DMA,
// K-loop unrolled x2 so the buffer index is a compile-time constant.
// ---------------------------------------------------------------------------
__global__ __launch_bounds__(256)
void gemm_f16_wmma(
    const _Float16* __restrict__ A, const _Float16* __restrict__ Wt,
    const float* __restrict__ bias, float* __restrict__ C)
{
    __shared__ _Float16 As[2][128 * LDSW_];   // 2 x 10 KB
    __shared__ _Float16 Bs[2][256 * LDSW_];   // 2 x 20 KB

    int tid  = threadIdx.x;
    int bn   = blockIdx.x * 256;
    int bm   = blockIdx.y * 128;
    int wave = tid >> 5;
    int wm   = wave & 1;      // 0..1 over M (64 rows each)
    int wn   = wave >> 1;     // 0..3 over N (64 cols each)

    // Staging split: A tile = 128 rows x 4 16B segments, B tile = 256 x 4.
    const _Float16* ga[2];
    const _Float16* gb[4];
    uint32_t lA[2][2], lB[2][4];              // [buf][chunk] LDS byte offsets
    #pragma unroll
    for (int q = 0; q < 2; ++q) {
        int c = tid + q * 256, row = c >> 2, seg = c & 3;
        ga[q]    = A + (size_t)(bm + row) * GK_ + seg * 8;
        int off  = row * LDSW_ + seg * 8;
        lA[0][q] = lds_off32(&As[0][off]);
        lA[1][q] = lds_off32(&As[1][off]);
    }
    #pragma unroll
    for (int q = 0; q < 4; ++q) {
        int c = tid + q * 256, row = c >> 2, seg = c & 3;
        gb[q]    = Wt + (size_t)(bn + row) * GK_ + seg * 8;
        int off  = row * LDSW_ + seg * 8;
        lB[0][q] = lds_off32(&Bs[0][off]);
        lB[1][q] = lds_off32(&Bs[1][off]);
    }

    v8f acc[4][4];
    #pragma unroll
    for (int i = 0; i < 4; ++i)
        #pragma unroll
        for (int j = 0; j < 4; ++j)
            #pragma unroll
            for (int v = 0; v < 8; ++v) acc[i][j][v] = 0.0f;

    auto refill = [&](int buf) {   // buf is a literal at every call site
        #pragma unroll
        for (int q = 0; q < 2; ++q) { async_g2l_b128(lA[buf][q], ga[q]); ga[q] += 32; }
        #pragma unroll
        for (int q = 0; q < 4; ++q) { async_g2l_b128(lB[buf][q], gb[q]); gb[q] += 32; }
    };
    auto burst = [&](int buf) {
        v16h bfr[4];
        #pragma unroll
        for (int j = 0; j < 4; ++j)
            bfr[j] = lds_frag(&Bs[buf][0], wn * 64 + 16 * j);
        #pragma unroll
        for (int i = 0; i < 4; ++i) {
            v16h af = lds_frag(&As[buf][0], wm * 64 + 16 * i);
            #pragma unroll
            for (int j = 0; j < 4; ++j)
                acc[i][j] = __builtin_amdgcn_wmma_f32_16x16x32_f16(
                    false, af, false, bfr[j], (short)0, acc[i][j], false, false);
        }
    };

    // prologue: async-fill buffer 0 (step 0) and buffer 1 (step 1)
    refill(0);
    refill(1);
    wait_async0();
    __syncthreads();

    // steady state, unrolled x2 (NSTEP_ = 64 is even)
    for (int kt = 0; kt < NSTEP_; kt += 2) {
        burst(0);                                   // step kt from buf 0
        {                                           // kt+1 <= 63 always valid
            wait_async0();                          // step kt+1 DMA complete
            __syncthreads();                        // buf0 reads done, data seen
            if (kt + 2 < NSTEP_) refill(0);         // step kt+2 -> buf 0
        }
        burst(1);                                   // step kt+1 from buf 1
        if (kt + 2 < NSTEP_) {
            wait_async0();                          // step kt+2 DMA complete
            __syncthreads();                        // buf1 reads done
            if (kt + 3 < NSTEP_) refill(1);         // step kt+3 -> buf 1
        }
    }

    // C/D layout: VGPR v -> M = v (lanes 0-15) / 8+v (lanes 16-31); N = lane&15
    int l    = tid & 31;
    int cn_l = l & 15;
    int mh   = (l >> 4) * 8;
    #pragma unroll
    for (int i = 0; i < 4; ++i)
        #pragma unroll
        for (int j = 0; j < 4; ++j) {
            int   cm = bm + wm * 64 + 16 * i + mh;
            int   cn = bn + wn * 64 + 16 * j + cn_l;
            float bv = bias[cn];
            #pragma unroll
            for (int v = 0; v < 8; ++v)
                C[(size_t)(cm + v) * GN_ + cn] = acc[i][j][v] + bv;
        }
}

// ---------------------------------------------------------------------------
// Fused: 8x8 windowed attention (Q,K,V rows are shared tokens), downsample,
// residual add, LayerNorm -> hn (f16) padded to (B, 64, 2048), pad rows = 0.
// grid = (64, B), block = 256 (8 waves). nsa = sum_j colsum[j] * V[j,:]
// ---------------------------------------------------------------------------
__global__ __launch_bounds__(256) void nsa_ds_ln(
    const float* __restrict__ Q, const float* __restrict__ Kt,
    const float* __restrict__ V, const float* __restrict__ fc,
    const float* __restrict__ dsw, const float* __restrict__ dsb,
    const float* __restrict__ lng, const float* __restrict__ lnb,
    _Float16* __restrict__ HN)
{
    int    w      = blockIdx.x;
    int    b      = blockIdx.y;
    int    tid    = threadIdx.x;
    size_t outRow = (size_t)(b * 64 + w) * 2048;

    if (w >= 57) {  // zero pad rows so downstream GEMM over 4096 rows is safe
        for (int d = tid; d < 2048; d += 256) HN[outRow + d] = (_Float16)0.0f;
        return;
    }

    __shared__ float attn_s[8][8];
    __shared__ float colsum_s[8];
    __shared__ float dsw_s[64];
    __shared__ float red1[8], red2[8];
    __shared__ float mu_s, rsig_s;

    int    wave  = tid >> 5;
    int    lane  = tid & 31;
    size_t tbase = (size_t)(b * 64 + w) * 2048;

    const float* qrow = Q + tbase + (size_t)wave * 2048;
    float dj[8];
    #pragma unroll
    for (int j = 0; j < 8; ++j) dj[j] = 0.0f;
    for (int d = lane; d < 2048; d += 32) {
        float qv = qrow[d];
        #pragma unroll
        for (int j = 0; j < 8; ++j)
            dj[j] += qv * Kt[tbase + (size_t)j * 2048 + d];
    }
    #pragma unroll
    for (int j = 0; j < 8; ++j) {
        float v = dj[j];
        for (int m = 16; m >= 1; m >>= 1) v += __shfl_xor(v, m, 32);
        dj[j] = v * SCALE_;
    }
    if (lane == 0) {
        float mx = dj[0];
        #pragma unroll
        for (int j = 1; j < 8; ++j) mx = fmaxf(mx, dj[j]);
        float e[8], s = 0.0f;
        #pragma unroll
        for (int j = 0; j < 8; ++j) { e[j] = __expf(dj[j] - mx); s += e[j]; }
        float inv = 1.0f / s;
        #pragma unroll
        for (int j = 0; j < 8; ++j) attn_s[wave][j] = e[j] * inv;
    }
    if (tid < 64) dsw_s[tid] = dsw[w * 64 + tid];
    __syncthreads();

    if (tid < 8) {
        float c = 0.0f;
        #pragma unroll
        for (int i = 0; i < 8; ++i) c += attn_s[i][tid];
        colsum_s[tid] = c;
    }
    __syncthreads();

    float hv[8];
    float dbv = dsb[w];
    #pragma unroll
    for (int kk = 0; kk < 8; ++kk) {
        int   d   = tid + kk * 256;
        float nsa = 0.0f;
        #pragma unroll
        for (int j = 0; j < 8; ++j)
            nsa += colsum_s[j] * V[tbase + (size_t)j * 2048 + d];
        float nw = 0.0f;
        for (int n = 0; n < 64; ++n)
            nw += dsw_s[n] * fc[(size_t)(b * 64 + n) * 2048 + d];
        hv[kk] = nw + dbv + nsa;
    }

    float s1 = 0.0f, s2 = 0.0f;
    #pragma unroll
    for (int kk = 0; kk < 8; ++kk) { s1 += hv[kk]; s2 += hv[kk] * hv[kk]; }
    for (int m = 16; m >= 1; m >>= 1) {
        s1 += __shfl_xor(s1, m, 32);
        s2 += __shfl_xor(s2, m, 32);
    }
    if (lane == 0) { red1[wave] = s1; red2[wave] = s2; }
    __syncthreads();
    if (tid == 0) {
        float a = 0.0f, c = 0.0f;
        #pragma unroll
        for (int q = 0; q < 8; ++q) { a += red1[q]; c += red2[q]; }
        float mu  = a * (1.0f / 2048.0f);
        float var = c * (1.0f / 2048.0f) - mu * mu;
        mu_s   = mu;
        rsig_s = rsqrtf(var + EPS_);
    }
    __syncthreads();
    float mu = mu_s, rs = rsig_s;
    #pragma unroll
    for (int kk = 0; kk < 8; ++kk) {
        int   d = tid + kk * 256;
        float x = (hv[kk] - mu) * rs * lng[d] + lnb[d];
        HN[outRow + d] = (_Float16)x;
    }
}

// ---------------------------------------------------------------------------
// Final attention per batch: S = (q2 k2^T)*scale via WMMA on a 64x64 padded
// tile (valid 57x57), softmax rows, column-sum, out = colsum . V2.
// ---------------------------------------------------------------------------
__global__ __launch_bounds__(256) void attn2_kernel(
    const _Float16* __restrict__ Qb, const _Float16* __restrict__ Kb,
    const float* __restrict__ V2, float* __restrict__ out)
{
    int b   = blockIdx.x;
    int tid = threadIdx.x;

    __shared__ float S[64][68];
    __shared__ float colsum[64];

    const _Float16* qbase = Qb + (size_t)b * 64 * 2048;
    const _Float16* kbase = Kb + (size_t)b * 64 * 2048;

    int wave = tid >> 5;
    int mi   = wave >> 1;
    int njb  = (wave & 1) * 2;

    v8f acc[2];
    #pragma unroll
    for (int j = 0; j < 2; ++j)
        #pragma unroll
        for (int v = 0; v < 8; ++v) acc[j][v] = 0.0f;

    for (int k0 = 0; k0 < 2048; k0 += 32) {
        v16h a = load_frag(qbase, 2048, mi * 16, k0);
        #pragma unroll
        for (int j = 0; j < 2; ++j) {
            v16h bb = load_frag(kbase, 2048, (njb + j) * 16, k0);
            acc[j] = __builtin_amdgcn_wmma_f32_16x16x32_f16(
                false, a, false, bb, (short)0, acc[j], false, false);
        }
    }

    int l    = tid & 31;
    int cn_l = l & 15;
    int mh   = (l >> 4) * 8;
    #pragma unroll
    for (int j = 0; j < 2; ++j)
        #pragma unroll
        for (int v = 0; v < 8; ++v)
            S[mi * 16 + mh + v][(njb + j) * 16 + cn_l] = acc[j][v] * SCALE_;
    __syncthreads();

    if (tid < 57) {
        int   i  = tid;
        float mx = -1e30f;
        for (int j = 0; j < 57; ++j) mx = fmaxf(mx, S[i][j]);
        float s = 0.0f;
        for (int j = 0; j < 57; ++j) {
            float e = __expf(S[i][j] - mx);
            S[i][j] = e;
            s += e;
        }
        float inv = 1.0f / s;
        for (int j = 0; j < 57; ++j) S[i][j] *= inv;
    }
    __syncthreads();
    if (tid < 57) {
        float c = 0.0f;
        for (int i = 0; i < 57; ++i) c += S[i][tid];
        colsum[tid] = c;
    }
    __syncthreads();

    #pragma unroll
    for (int kk = 0; kk < 8; ++kk) {
        int   d = tid + kk * 256;
        float o = 0.0f;
        for (int j = 0; j < 57; ++j)
            o += colsum[j] * V2[(size_t)(b * 64 + j) * 2048 + d];
        out[(size_t)b * 2048 + d] = o;
    }
}

// ---------------------------------------------------------------------------
extern "C" void kernel_launch(void* const* d_in, const int* in_sizes, int n_in,
                              void* d_out, int out_size, void* d_ws, size_t ws_size,
                              hipStream_t stream)
{
    (void)in_sizes; (void)n_in; (void)out_size; (void)ws_size;

    const float* fc     = (const float*)d_in[0];
    const float* nsa_wq = (const float*)d_in[1];
    const float* nsa_bq = (const float*)d_in[2];
    const float* nsa_wk = (const float*)d_in[3];
    const float* nsa_bk = (const float*)d_in[4];
    const float* nsa_wv = (const float*)d_in[5];
    const float* nsa_bv = (const float*)d_in[6];
    const float* ds_w   = (const float*)d_in[7];
    const float* ds_b   = (const float*)d_in[8];
    const float* ln_g   = (const float*)d_in[9];
    const float* ln_b   = (const float*)d_in[10];
    const float* sa_wq  = (const float*)d_in[11];
    const float* sa_bq  = (const float*)d_in[12];
    const float* sa_wk  = (const float*)d_in[13];
    const float* sa_bk  = (const float*)d_in[14];
    const float* sa_wv  = (const float*)d_in[15];
    const float* sa_bv  = (const float*)d_in[16];
    float* out = (float*)d_out;

    char*        ws  = (char*)d_ws;
    const size_t MB  = 1024ull * 1024ull;
    _Float16* XBF  = (_Float16*)(ws + 0);        // 16 MB: fc in f16 (4096x2048)
    _Float16* WBF0 = (_Float16*)(ws + 16 * MB);  //  8 MB each: weights f16
    _Float16* WBF1 = (_Float16*)(ws + 24 * MB);
    _Float16* WBF2 = (_Float16*)(ws + 32 * MB);
    float*    Qf   = (float*)(ws + 40 * MB);     // 32 MB each
    float*    Kf   = (float*)(ws + 72 * MB);
    float*    Vf   = (float*)(ws + 104 * MB);
    _Float16* HN   = (_Float16*)(ws + 136 * MB); // 16 MB: hn f16 padded (B,64,D)
    _Float16* Q2BF = XBF;                        // reuse after stage-2 GEMMs
    _Float16* K2BF = (_Float16*)(ws + 16 * MB);  // 16 MB spans WBF0+WBF1

    const int NX = 64 * 64 * 2048;
    const int NW = 2048 * 2048;
    dim3 blk(256);
    dim3 gX((NX + 255) / 256);
    dim3 gW((NW + 255) / 256);
    dim3 ggrid(GN_ / 256, GM_ / 128);            // (8, 32)

    // Stage 0: down-convert activations + NSA weights
    cvt_f16<<<gX, blk, 0, stream>>>(fc,     XBF,  NX);
    cvt_f16<<<gW, blk, 0, stream>>>(nsa_wq, WBF0, NW);
    cvt_f16<<<gW, blk, 0, stream>>>(nsa_wk, WBF1, NW);
    cvt_f16<<<gW, blk, 0, stream>>>(nsa_wv, WBF2, NW);

    // Stage 1: token-level Q/K/V projections (windows share tokens)
    gemm_f16_wmma<<<ggrid, blk, 0, stream>>>(XBF, WBF0, nsa_bq, Qf);
    gemm_f16_wmma<<<ggrid, blk, 0, stream>>>(XBF, WBF1, nsa_bk, Kf);
    gemm_f16_wmma<<<ggrid, blk, 0, stream>>>(XBF, WBF2, nsa_bv, Vf);

    // Stage 2: window attention + downsample + LN -> hn (f16, zero-padded)
    nsa_ds_ln<<<dim3(64, 64), blk, 0, stream>>>(Qf, Kf, Vf, fc, ds_w, ds_b,
                                                ln_g, ln_b, HN);

    // Stage 3: second projections
    cvt_f16<<<gW, blk, 0, stream>>>(sa_wq, WBF0, NW);
    cvt_f16<<<gW, blk, 0, stream>>>(sa_wk, WBF1, NW);
    cvt_f16<<<gW, blk, 0, stream>>>(sa_wv, WBF2, NW);
    gemm_f16_wmma<<<ggrid, blk, 0, stream>>>(HN, WBF0, sa_bq, Qf);
    gemm_f16_wmma<<<ggrid, blk, 0, stream>>>(HN, WBF1, sa_bk, Kf);
    gemm_f16_wmma<<<ggrid, blk, 0, stream>>>(HN, WBF2, sa_bv, Vf);

    // Stage 4: final attention (WMMA scores) -> out (B, D)
    cvt_f16<<<gX, blk, 0, stream>>>(Qf, Q2BF, NX);
    cvt_f16<<<gX, blk, 0, stream>>>(Kf, K2BF, NX);
    attn2_kernel<<<dim3(64), blk, 0, stream>>>(Q2BF, K2BF, Vf, out);
}
